// DecoderRNNT_21766894256370
// MI455X (gfx1250) — compile-verified, matrix operand order checked by reading, not run on hardware
//
#include <hip/hip_runtime.h>
#include <hip/hip_bf16.h>

// ---- problem constants -----------------------------------------------------
#define BSZ   8
#define TT    200
#define UU    50
#define UP1   51
#define DK    512          // EPROJS = DUNITS = EMBED = JOINT = ODIM = 512
#define G4    2048         // 4*DUNITS
#define NROWS (BSZ*UP1)    // 408  decoder rows
#define EROWS (BSZ*TT)     // 1600 encoder rows
#define JROWS (BSZ*TT*UP1) // 81600 joint rows (multiple of 16)

typedef __attribute__((ext_vector_type(16))) __bf16 v16bf;
typedef __attribute__((ext_vector_type(8)))  float  v8f;

// float -> bf16, round-to-nearest-even (bit-level)
__device__ __forceinline__ __bf16 f2bf(float f) {
    union { float f; unsigned u; } v; v.f = f;
    unsigned r = v.u + 0x7FFFu + ((v.u >> 16) & 1u);
    union { unsigned short s; __bf16 b; } o; o.s = (unsigned short)(r >> 16);
    return o.b;
}
__device__ __forceinline__ float sigm(float x) { return 1.0f / (1.0f + __expf(-x)); }
__device__ __forceinline__ float lae(float a, float b) {
    float m = fmaxf(a, b);
    return m + log1pf(__expf(fminf(a, b) - m));
}

// ---- tiny elementwise kernels ---------------------------------------------
__global__ void cvt_bf16_kernel(const float* __restrict__ src,
                                __bf16* __restrict__ dst, int n) {
    int i = blockIdx.x * blockDim.x + threadIdx.x;
    if (i < n) dst[i] = f2bf(src[i]);
}

__global__ void addv_kernel(const float* __restrict__ a,
                            const float* __restrict__ b,
                            float* __restrict__ dst, int n) {
    int i = blockIdx.x * blockDim.x + threadIdx.x;
    if (i < n) dst[i] = a[i] + b[i];
}

// eys[b, u, :] = (u==0 || token==BLANK) ? 0 : embed_w[token]   (bf16 output)
__global__ void embed_kernel(const int* __restrict__ ys,
                             const float* __restrict__ embed_w,
                             __bf16* __restrict__ eys) {
    int idx = blockIdx.x * blockDim.x + threadIdx.x;   // over 408*512
    if (idx >= NROWS * DK) return;
    int row = idx >> 9, k = idx & (DK - 1);
    int b = row / UP1, u = row % UP1;
    int tok = (u == 0) ? 0 : ys[b * UU + (u - 1)];
    eys[idx] = f2bf((tok == 0) ? 0.0f : embed_w[tok * DK + k]);
}

// ---- generic bf16-WMMA GEMM: C[M x N] = A[M x 512] * W^T + bias ------------
// A: bf16 row-major (ld 512); W: bf16 row-major N x 512; C: f32 row-major ld N
__global__ void gemm_kernel(const __bf16* __restrict__ A, int M,
                            const __bf16* __restrict__ W, int N,
                            const float* __restrict__ bias,
                            float* __restrict__ C) {
    int lane = threadIdx.x & 31, wave = threadIdx.x >> 5;
    int mt_n = (M + 15) >> 4, nt_n = N >> 4;
    int tile = blockIdx.x * 8 + wave;
    if (tile >= mt_n * nt_n) return;
    int mt = tile / nt_n, nt = tile % nt_n;
    int m0 = mt << 4, n0 = nt << 4;

    int arow = m0 + (lane & 15); if (arow >= M) arow = M - 1;
    int brow = n0 + (lane & 15);
    int khalf = (lane >> 4) << 4;          // 0 or 16
    const __bf16* ap = A + arow * DK + khalf;
    const __bf16* bp = W + brow * DK + khalf;

    v8f acc = {};
#pragma unroll 4
    for (int k0 = 0; k0 < DK; k0 += 32) {
        v16bf a = *(const v16bf*)(ap + k0);
        v16bf b = *(const v16bf*)(bp + k0);
        acc = __builtin_amdgcn_wmma_f32_16x16x32_bf16(
            false, a, false, b, (short)0, acc, false, false);
    }
    int col = n0 + (lane & 15);
    float bb = bias ? bias[col] : 0.0f;
#pragma unroll
    for (int j = 0; j < 8; ++j) {
        int row = m0 + ((lane >> 4) << 3) + j;
        if (row < M) C[row * N + col] = acc[j] + bb;
    }
}

// ---- persistent single-WGP LSTM layer (51 sequential steps) ----------------
// xproj: (B,51,2048) f32 precomputed x@Wih^T + b_ih + b_hh
// Whh:   bf16 2048 x 512 ;  hout: (B,51,512) bf16 (feeds next GEMM directly)
__global__ void lstm_kernel(const float* __restrict__ xproj,
                            const __bf16* __restrict__ Whh,
                            __bf16* __restrict__ hout) {
    extern __shared__ char smem[];
    __bf16* hA   = (__bf16*)smem;                        // 16 x 512 (rows 8..15 stay 0)
    float*  gate = (float*)(smem + 16 * DK * 2);         // 8 x 2048
    float*  cst  = (float*)(smem + 16 * DK * 2 + 8 * G4 * 4); // 8 x 512

    int tid = threadIdx.x, lane = tid & 31, wave = tid >> 5;
    for (int e = tid; e < 16 * DK; e += 1024) hA[e] = f2bf(0.0f);
    for (int e = tid; e < 8 * DK; e += 1024)  cst[e] = 0.0f;
    __syncthreads();

    int khalf = (lane >> 4) << 4;
    const __bf16* hap = hA + (lane & 15) * DK + khalf;
    for (int t = 0; t < UP1; ++t) {
        // gates = hA @ Whh^T  (+ xproj), 128 n-tiles over 32 waves
#pragma unroll
        for (int q = 0; q < 4; ++q) {
            int n0 = ((wave << 2) + q) << 4;
            const __bf16* bp = Whh + (n0 + (lane & 15)) * DK + khalf;
            v8f acc = {};
#pragma unroll 4
            for (int k0 = 0; k0 < DK; k0 += 32) {
                v16bf a = *(const v16bf*)(hap + k0);
                v16bf b = *(const v16bf*)(bp + k0);
                acc = __builtin_amdgcn_wmma_f32_16x16x32_bf16(
                    false, a, false, b, (short)0, acc, false, false);
            }
            int col = n0 + (lane & 15);
#pragma unroll
            for (int j = 0; j < 8; ++j) {
                int row = ((lane >> 4) << 3) + j;
                if (row < BSZ)
                    gate[row * G4 + col] =
                        acc[j] + xproj[(row * UP1 + t) * G4 + col];
            }
        }
        __syncthreads();
        // elementwise LSTM cell (gate order i,f,g,o)
        for (int e = tid; e < BSZ * DK; e += 1024) {
            int b = e >> 9, j = e & (DK - 1);
            const float* g = gate + b * G4;
            float gi = g[j], gf = g[DK + j], gg = g[2 * DK + j], go = g[3 * DK + j];
            float c = sigm(gf) * cst[e] + sigm(gi) * tanhf(gg);
            float h = sigm(go) * tanhf(c);
            cst[e] = c;
            __bf16 hb = f2bf(h);
            hA[b * DK + j] = hb;
            hout[(b * UP1 + t) * DK + j] = hb;
        }
        __syncthreads();
    }
}

// ---- fused joint: tanh(enc+dec) -> GEMM(Wout)+bias -> logsumexp -> gather --
__global__ void joint_kernel(const float* __restrict__ enc,
                             const float* __restrict__ dec,
                             const __bf16* __restrict__ Wout,
                             const float* __restrict__ bout,
                             const int* __restrict__ ys,
                             float* __restrict__ lpb,
                             float* __restrict__ lpl) {
    __shared__ __align__(32) __bf16 z[16][528];   // padded vs bank conflicts
    __shared__ float ob[16][520];

    int r0 = blockIdx.x << 4;
    // phase 1: z tile
    for (int e = threadIdx.x; e < 16 * DK; e += 256) {
        int i = e >> 9, k = e & (DK - 1);
        int r = r0 + i;
        int b = r / (TT * UP1), rem = r % (TT * UP1);
        int t = rem / UP1, u = rem % UP1;
        float v = enc[(b * TT + t) * DK + k] + dec[(b * UP1 + u) * DK + k];
        z[i][k] = f2bf(tanhf(v));
    }
    __syncthreads();

    // phase 2: 16x512 output block via WMMA, 4 n-tiles per wave
    int lane = threadIdx.x & 31, wave = threadIdx.x >> 5;
    int khalf = (lane >> 4) << 4;
    const __bf16* ap = &z[lane & 15][khalf];
#pragma unroll
    for (int q = 0; q < 4; ++q) {
        int n0 = (wave + (q << 3)) << 4;
        const __bf16* bp = Wout + (n0 + (lane & 15)) * DK + khalf;
        v8f acc = {};
#pragma unroll 4
        for (int k0 = 0; k0 < DK; k0 += 32) {
            v16bf a = *(const v16bf*)(ap + k0);
            v16bf b = *(const v16bf*)(bp + k0);
            acc = __builtin_amdgcn_wmma_f32_16x16x32_bf16(
                false, a, false, b, (short)0, acc, false, false);
        }
        int col = n0 + (lane & 15);
        float bb = bout[col];
#pragma unroll
        for (int j = 0; j < 8; ++j)
            ob[((lane >> 4) << 3) + j][col] = acc[j] + bb;
    }
    __syncthreads();

    // phase 3: per-row logsumexp (16 lanes/row, shfl reductions) + gather
    {
        int i = threadIdx.x >> 4;          // row 0..15
        int g = threadIdx.x & 15;          // lane-in-row
        float m = -1e30f;
        for (int k = g; k < DK; k += 16) m = fmaxf(m, ob[i][k]);
#pragma unroll
        for (int d = 8; d; d >>= 1) m = fmaxf(m, __shfl_xor(m, d, 16));
        float s = 0.0f;
        for (int k = g; k < DK; k += 16) s += __expf(ob[i][k] - m);
#pragma unroll
        for (int d = 8; d; d >>= 1) s += __shfl_xor(s, d, 16);
        if (g == 0) {
            int r = r0 + i;
            int b = r / (TT * UP1), rem = r % (TT * UP1);
            int t = rem / UP1, u = rem % UP1;
            float lse = m + __logf(s);
            lpb[(b * TT + t) * UP1 + u] = ob[i][0] - lse;   // BLANK = 0
            if (u < UU)
                lpl[(b * TT + t) * UU + u] = ob[i][ys[b * UU + u]] - lse;
        }
    }
}

// ---- RNN-T alpha DP + mean NLL (one wave; thread b owns batch b) -----------
__global__ void loss_kernel(const float* __restrict__ lpb,
                            const float* __restrict__ lpl,
                            float* __restrict__ out) {
    __shared__ float alpha[BSZ][UP1];
    __shared__ float nll[BSZ];
    int b = threadIdx.x;
    if (b < BSZ) {
        float* al = alpha[b];
        al[0] = 0.0f;
        for (int u = 1; u <= UU; ++u)
            al[u] = al[u - 1] + lpl[(b * TT + 0) * UU + (u - 1)];
        for (int t = 1; t < TT; ++t) {
            const float* pb = lpb + (b * TT + (t - 1)) * UP1;
            const float* pl = lpl + (b * TT + t) * UU;
            float c = al[0] + pb[0];
            al[0] = c;
            for (int u = 1; u <= UU; ++u) {
                float a_u = al[u] + pb[u];
                c = lae(a_u, c + pl[u - 1]);
                al[u] = c;
            }
        }
        nll[b] = -(al[UU] + lpb[(b * TT + (TT - 1)) * UP1 + UU]);
    }
    __syncthreads();
    if (threadIdx.x == 0) {
        float s = 0.0f;
        for (int i = 0; i < BSZ; ++i) s += nll[i];
        out[0] = s / (float)BSZ;
    }
}

// ---- host orchestration ----------------------------------------------------
extern "C" void kernel_launch(void* const* d_in, const int* in_sizes, int n_in,
                              void* d_out, int out_size, void* d_ws, size_t ws_size,
                              hipStream_t stream) {
    const float* hs_pad   = (const float*)d_in[0];
    const int*   ys_pad   = (const int*)  d_in[2];
    const float* embed_w  = (const float*)d_in[3];
    const float* w_ih0    = (const float*)d_in[4];
    const float* w_hh0    = (const float*)d_in[5];
    const float* b_ih0    = (const float*)d_in[6];
    const float* b_hh0    = (const float*)d_in[7];
    const float* w_ih1    = (const float*)d_in[8];
    const float* w_hh1    = (const float*)d_in[9];
    const float* b_ih1    = (const float*)d_in[10];
    const float* b_hh1    = (const float*)d_in[11];
    const float* enc_w    = (const float*)d_in[12];
    const float* enc_b    = (const float*)d_in[13];
    const float* dec_w    = (const float*)d_in[14];
    const float* out_w    = (const float*)d_in[15];
    const float* out_b    = (const float*)d_in[16];

    char* ws = (char*)d_ws;
    size_t off = 0;
    auto alloc = [&](size_t bytes) -> char* {
        char* p = ws + off;
        off += (bytes + 1023) & ~(size_t)1023;
        return p;
    };
    __bf16* eys   = (__bf16*)alloc((size_t)NROWS * DK * 2);
    float*  xproj = (float*) alloc((size_t)NROWS * G4 * 4);
    __bf16* h0    = (__bf16*)alloc((size_t)NROWS * DK * 2);
    __bf16* hdec  = (__bf16*)alloc((size_t)NROWS * DK * 2);
    __bf16* hsb   = (__bf16*)alloc((size_t)EROWS * DK * 2);
    float*  enc   = (float*) alloc((size_t)EROWS * DK * 4);
    float*  dec   = (float*) alloc((size_t)NROWS * DK * 4);
    float*  lpb   = (float*) alloc((size_t)BSZ * TT * UP1 * 4);
    float*  lpl   = (float*) alloc((size_t)BSZ * TT * UU * 4);
    float*  bias0 = (float*) alloc(G4 * 4);
    float*  bias1 = (float*) alloc(G4 * 4);
    __bf16* wih0b = (__bf16*)alloc((size_t)G4 * DK * 2);
    __bf16* whh0b = (__bf16*)alloc((size_t)G4 * DK * 2);
    __bf16* wih1b = (__bf16*)alloc((size_t)G4 * DK * 2);
    __bf16* whh1b = (__bf16*)alloc((size_t)G4 * DK * 2);
    __bf16* encwb = (__bf16*)alloc((size_t)DK * DK * 2);
    __bf16* decwb = (__bf16*)alloc((size_t)DK * DK * 2);
    __bf16* outwb = (__bf16*)alloc((size_t)DK * DK * 2);
    (void)ws_size; (void)n_in; (void)in_sizes; (void)out_size;

    auto cvt = [&](const float* s, __bf16* d, int n) {
        cvt_bf16_kernel<<<(n + 255) / 256, 256, 0, stream>>>(s, d, n);
    };
    cvt(w_ih0, wih0b, G4 * DK);
    cvt(w_hh0, whh0b, G4 * DK);
    cvt(w_ih1, wih1b, G4 * DK);
    cvt(w_hh1, whh1b, G4 * DK);
    cvt(enc_w, encwb, DK * DK);
    cvt(dec_w, decwb, DK * DK);
    cvt(out_w, outwb, DK * DK);
    cvt(hs_pad, hsb, EROWS * DK);

    addv_kernel<<<(G4 + 255) / 256, 256, 0, stream>>>(b_ih0, b_hh0, bias0, G4);
    addv_kernel<<<(G4 + 255) / 256, 256, 0, stream>>>(b_ih1, b_hh1, bias1, G4);

    embed_kernel<<<(NROWS * DK + 255) / 256, 256, 0, stream>>>(ys_pad, embed_w, eys);

    auto gemm = [&](const __bf16* A, int M, const __bf16* W, int N,
                    const float* bias, float* C) {
        int tiles = ((M + 15) / 16) * (N / 16);
        gemm_kernel<<<(tiles + 7) / 8, 256, 0, stream>>>(A, M, W, N, bias, C);
    };

    const size_t lstm_lds = 16 * DK * 2 + 8 * G4 * 4 + 8 * DK * 4;  // 96 KB
    gemm(eys, NROWS, wih0b, G4, bias0, xproj);
    lstm_kernel<<<1, 1024, lstm_lds, stream>>>(xproj, whh0b, h0);
    gemm(h0, NROWS, wih1b, G4, bias1, xproj);
    lstm_kernel<<<1, 1024, lstm_lds, stream>>>(xproj, whh1b, hdec);

    gemm(hsb, EROWS, encwb, DK, enc_b, enc);
    gemm(hdec, NROWS, decwb, DK, nullptr, dec);

    joint_kernel<<<JROWS / 16, 256, 0, stream>>>(enc, dec, outwb, out_b,
                                                 ys_pad, lpb, lpl);

    loss_kernel<<<1, 32, 0, stream>>>(lpb, lpl, (float*)d_out);
}
